// DoomLiquidNet_7404523618770
// MI455X (gfx1250) — matrix operand order repaired and checked by reading, not compiled, not run on hardware
//
#include <hip/hip_runtime.h>
#include <hip/hip_bf16.h>
#include <math.h>

typedef __attribute__((ext_vector_type(16))) _Float16 v16h;
typedef __attribute__((ext_vector_type(8)))  _Float16 v8h;
typedef __attribute__((ext_vector_type(2)))  _Float16 v2h;
typedef __attribute__((ext_vector_type(8)))  float    v8f;

#define BATCH 8
#define TSTEPS 256
#define FRAMES 2048        // B*T
#define FEAT 12544         // 64*14*14
#define BBU 128
#define UNITS 64
#define NACT 8

__device__ inline v16h mk16(v8h lo, v8h hi) {
  return __builtin_shufflevector(lo, hi, 0, 1, 2, 3, 4, 5, 6, 7,
                                 8, 9, 10, 11, 12, 13, 14, 15);
}

// ---------------------------------------------------------------------------
// Kernel 0: weight prep.
//  - bb_w[:FEAT] (K-major f32 [12544][128]) -> N-major f16 [128][12544]
//    so GEMM B-fragments are one contiguous 32-B load per lane.
//  - conv2_w [64][32][4][4] f32 -> f16 [64][512] (already im2col K order).
//  - head weights [128][64] f32 -> transposed [4][64][128] f32 so the scan's
//    per-thread dot products read contiguous 512-B rows.
// ---------------------------------------------------------------------------
__global__ void prep_weights_kernel(const float* __restrict__ bb_w,
                                    _Float16* __restrict__ bbwT,
                                    const float* __restrict__ w2f,
                                    _Float16* __restrict__ w2h,
                                    const float* __restrict__ ff1_w,
                                    const float* __restrict__ ff2_w,
                                    const float* __restrict__ ta_w,
                                    const float* __restrict__ tb_w,
                                    float* __restrict__ whT) {
  int idx = blockIdx.x * blockDim.x + threadIdx.x;
  if (idx < FEAT * BBU) {
    int k = idx / BBU, n = idx % BBU;
    bbwT[(size_t)n * FEAT + k] = (_Float16)bb_w[idx];
  }
  if (idx < 64 * 512) w2h[idx] = (_Float16)w2f[idx];
  if (idx < 4 * UNITS * BBU) {               // 32768
    int h = idx >> 13, rem = idx & 8191, u = rem >> 7, kk = rem & 127;
    const float* src = (h == 0) ? ff1_w : (h == 1) ? ff2_w
                       : (h == 2) ? ta_w : tb_w;
    whT[idx] = src[kk * UNITS + u];
  }
}

// ---------------------------------------------------------------------------
// Kernel 1: fused conv1 + conv2 per frame. One workgroup (128 thr = 4 waves)
// per frame. conv1 (f32 VALU, relu) -> LDS f16 [32][30][30]; conv2 as implicit
// GEMM on v_wmma_f32_16x16x32_f16: M=64 oc (1 m-tile/wave), N=196 px
// (13 n-tiles, clamped — garbage columns never stored), K=512 (16x32).
// B-fragment = 8 straight-line ds_load_b32 (one channel, 4 rows x 4 halves).
// ---------------------------------------------------------------------------
__global__ __launch_bounds__(128) void conv_frame_kernel(
    const float* __restrict__ x, const float* __restrict__ w1,
    const float* __restrict__ b1, const _Float16* __restrict__ w2,
    const float* __restrict__ b2, _Float16* __restrict__ r_in) {
  __shared__ _Float16 s_c1[32 * 30 * 30];  // 57,600 B
  const int frame = blockIdx.x;
  const int tid = threadIdx.x;
  const float* xf = x + (size_t)frame * (3 * 62 * 62);

  // conv1: 28800 outputs / 128 threads = 225 each
  for (int o = tid; o < 28800; o += 128) {
    int oc = o / 900, p = o % 900, oh = p / 30, ow = p % 30;
    float acc = b1[oc];
    const float* wrow = w1 + oc * 48;
#pragma unroll
    for (int ic = 0; ic < 3; ++ic)
#pragma unroll
      for (int kh = 0; kh < 4; ++kh) {
        const float* xr = xf + ic * 3844 + (oh * 2 + kh) * 62 + ow * 2;
        const float* wr = wrow + ic * 16 + kh * 4;
#pragma unroll
        for (int kw = 0; kw < 4; ++kw) acc += xr[kw] * wr[kw];
      }
    s_c1[o] = (_Float16)fmaxf(acc, 0.0f);
  }
  __syncthreads();

  // conv2 WMMA
  const int lane = tid & 31;
  const int mtile = tid >> 5;
  const int l16 = lane & 15;
  const int hi = (lane >= 16) ? 1 : 0;
  const _Float16* __restrict__ wrow = w2 + (mtile * 16 + l16) * 512;
  const int aoff0 = hi ? 16 : 0;            // A K-group offset (contiguous k = 16h + s)

  for (int nt = 0; nt < 13; ++nt) {
    const int pcol = nt * 16 + l16;          // D column (output pixel)
    const int pc = (pcol < 196) ? pcol : 195;  // clamp: no divergence in gather
    const int oh = pc / 14, ow = pc % 14;
    const _Float16* __restrict__ gb = s_c1 + oh * 60 + ow * 2;
    v8f acc = {};
#pragma unroll 4
    for (int kc = 0; kc < 16; ++kc) {
      const int k0 = kc * 32;
      // A fragment: two contiguous 8-half groups from global f16 weights
      v8h alo = *(const v8h*)(wrow + k0 + aoff0);
      v8h ahi = *(const v8h*)(wrow + k0 + aoff0 + 8);
      v16h a = mk16(alo, ahi);
      // B fragment: channel c = 2*kc + hi; r = kh*4+kw -> element index
      const _Float16* __restrict__ cb = gb + (2 * kc + hi) * 900;
      v16h b;
#pragma unroll
      for (int kh = 0; kh < 4; ++kh) {
        v2h p0 = *(const v2h*)(cb + kh * 30);
        v2h p1 = *(const v2h*)(cb + kh * 30 + 2);
        b[kh * 4 + 0] = p0[0]; b[kh * 4 + 1] = p0[1];
        b[kh * 4 + 2] = p1[0]; b[kh * 4 + 3] = p1[1];
      }
      acc = __builtin_amdgcn_wmma_f32_16x16x32_f16(false, a, false, b,
                                                   (short)0, acc, false, false);
    }
    if (pcol < 196) {
#pragma unroll
      for (int i = 0; i < 8; ++i) {
        int oc = mtile * 16 + i + hi * 8;
        float v = acc[i] + b2[oc];
        r_in[(size_t)frame * FEAT + oc * 196 + pcol] = (_Float16)fmaxf(v, 0.0f);
      }
    }
  }
}

// ---------------------------------------------------------------------------
// Kernel 2: hoisted backbone GEMM  xz[2048][128] = r_in(f16) @ bb_w[:FEAT](f16)
// + bb_b.  256 thr = 8 waves; workgroup owns one 16-row M-tile; wave w owns
// N-tile w. K = 392 chunks of 32, unrolled x2. Operands L2-resident
// (51 MB + 3.2 MB << 192 MB L2).
// ---------------------------------------------------------------------------
__global__ __launch_bounds__(256) void gemm_xz_kernel(
    const _Float16* __restrict__ rin, const _Float16* __restrict__ bbwT,
    const float* __restrict__ bb_b, float* __restrict__ xz) {
  const int tid = threadIdx.x, lane = tid & 31, wave = tid >> 5;
  const int l16 = lane & 15;
  const int hi = (lane >= 16) ? 1 : 0;
  const int mrow = blockIdx.x * 16 + l16;   // A row
  const int ncol = wave * 16 + l16;         // B col
  const _Float16* __restrict__ ap = rin + (size_t)mrow * FEAT;
  const _Float16* __restrict__ bp = bbwT + (size_t)ncol * FEAT + (hi ? 16 : 0);
  const int aoff0 = hi ? 16 : 0;
  v8f acc = {};
#pragma unroll 2
  for (int kc = 0; kc < 392; ++kc) {
    const int k0 = kc * 32;
    __builtin_prefetch(ap + k0 + 1024, 0, 1);   // global_prefetch_b8
    v8h alo = *(const v8h*)(ap + k0 + aoff0);
    v8h ahi = *(const v8h*)(ap + k0 + aoff0 + 8);
    v16h a = mk16(alo, ahi);
    v16h b = *(const v16h*)(bp + k0);           // contiguous 32 B per lane
    acc = __builtin_amdgcn_wmma_f32_16x16x32_f16(false, a, false, b,
                                                 (short)0, acc, false, false);
  }
  const float bias = bb_b[ncol];
#pragma unroll
  for (int i = 0; i < 8; ++i) {
    int M = blockIdx.x * 16 + i + hi * 8;
    xz[(size_t)M * BBU + ncol] = acc[i] + bias;
  }
}

// ---------------------------------------------------------------------------
// Kernel 3: CfC scan. One workgroup per batch element (independent h state).
// 128 threads. W_hh (32 KB) in LDS; transposed head weights streamed from L2
// with contiguous float4 rows. f32 recurrence (latency-bound critical path).
// ---------------------------------------------------------------------------
__global__ __launch_bounds__(128) void cfc_scan_kernel(
    const float* __restrict__ xz, const float* __restrict__ bb_w,
    const float* __restrict__ whT,
    const float* __restrict__ ff1_b, const float* __restrict__ ff2_b,
    const float* __restrict__ ta_b, const float* __restrict__ tb_b,
    const float* __restrict__ out_w, const float* __restrict__ out_b,
    float* __restrict__ logits, float* __restrict__ hx_out) {
  __shared__ float s_whh[UNITS * BBU];  // 32 KB
  __shared__ float s_h[UNITS];
  __shared__ float s_z[BBU];
  __shared__ float s_heads[4 * UNITS];
  const int b = blockIdx.x, tid = threadIdx.x;

  for (int i = tid; i < UNITS * BBU; i += 128)
    s_whh[i] = bb_w[(size_t)FEAT * BBU + i];
  if (tid < UNITS) s_h[tid] = 0.0f;

  // per-thread head assignment (2 outputs: flat tid and tid+128)
  const int head0 = tid >> 6, u0 = tid & 63;          // heads 0..1
  const int head1 = head0 + 2;                         // heads 2..3
  const float4* __restrict__ w0 = (const float4*)(whT + head0 * 8192 + u0 * 128);
  const float4* __restrict__ w1 = (const float4*)(whT + head1 * 8192 + u0 * 128);
  const float hb0 = (head0 == 0) ? ff1_b[u0] : ff2_b[u0];
  const float hb1 = (head1 == 2) ? ta_b[u0] : tb_b[u0];
  __syncthreads();

  for (int t = 0; t < TSTEPS; ++t) {
    // z = lecun_tanh(xz + h @ W_hh)   (xz already holds x@W + bb_b)
    float acc = xz[((size_t)b * TSTEPS + t) * BBU + tid];
#pragma unroll 8
    for (int k = 0; k < UNITS; ++k) acc += s_h[k] * s_whh[k * BBU + tid];
    s_z[tid] = 1.7159f * tanhf(0.666f * acc);
    __syncthreads();
    // four heads: 256 dot-128 outputs, 2 per thread, contiguous weight rows
    float d0 = hb0, d1 = hb1;
#pragma unroll 8
    for (int kk = 0; kk < BBU / 4; ++kk) {
      float4 q0 = w0[kk], q1 = w1[kk];
      float z0 = s_z[4 * kk], z1 = s_z[4 * kk + 1];
      float z2 = s_z[4 * kk + 2], z3 = s_z[4 * kk + 3];
      d0 += z0 * q0.x + z1 * q0.y + z2 * q0.z + z3 * q0.w;
      d1 += z0 * q1.x + z1 * q1.y + z2 * q1.z + z3 * q1.w;
    }
    s_heads[tid] = tanhf(d0);          // heads 0,1: ff1/ff2 get tanh
    s_heads[tid + 128] = d1;           // heads 2,3: ta/tb raw
    __syncthreads();
    if (tid < UNITS) {
      float ti = 1.0f / (1.0f + expf(-(s_heads[128 + tid] + s_heads[192 + tid])));
      s_h[tid] = s_heads[tid] * (1.0f - ti) + ti * s_heads[64 + tid];
    }
    __syncthreads();
    if (tid < NACT) {
      float lg = out_b[tid];
#pragma unroll 8
      for (int u = 0; u < UNITS; ++u) lg += s_h[u] * out_w[u * NACT + tid];
      logits[((size_t)b * TSTEPS + t) * NACT + tid] = lg;
    }
    __syncthreads();
  }
  if (tid < UNITS) hx_out[b * UNITS + tid] = s_h[tid];
}

// ---------------------------------------------------------------------------
extern "C" void kernel_launch(void* const* d_in, const int* in_sizes, int n_in,
                              void* d_out, int out_size, void* d_ws,
                              size_t ws_size, hipStream_t stream) {
  const float* x       = (const float*)d_in[0];
  const float* conv1_w = (const float*)d_in[1];
  const float* conv1_b = (const float*)d_in[2];
  const float* conv2_w = (const float*)d_in[3];
  const float* conv2_b = (const float*)d_in[4];
  const float* bb_w    = (const float*)d_in[5];
  const float* bb_b    = (const float*)d_in[6];
  const float* ff1_w   = (const float*)d_in[7];
  const float* ff1_b   = (const float*)d_in[8];
  const float* ff2_w   = (const float*)d_in[9];
  const float* ff2_b   = (const float*)d_in[10];
  const float* ta_w    = (const float*)d_in[11];
  const float* ta_b    = (const float*)d_in[12];
  const float* tb_w    = (const float*)d_in[13];
  const float* tb_b    = (const float*)d_in[14];
  const float* out_w   = (const float*)d_in[15];
  const float* out_b   = (const float*)d_in[16];

  // workspace layout (256-B aligned offsets)
  char* ws = (char*)d_ws;
  _Float16* r_in = (_Float16*)ws;                    // 2048*12544*2 = 51,380,224 B
  _Float16* bbwT = (_Float16*)(ws + 51380224);       // 128*12544*2 =  3,211,264 B
  _Float16* w2h  = (_Float16*)(ws + 54591488);       // 64*512*2    =     65,536 B
  float*    xz   = (float*)   (ws + 54657024);       // 2048*128*4  =  1,048,576 B
  float*    whT  = (float*)   (ws + 55705600);       // 4*64*128*4  =    131,072 B

  float* logits = (float*)d_out;                       // [8][256][8]
  float* hx_out = logits + (size_t)BATCH * TSTEPS * NACT;  // [8][64]

  prep_weights_kernel<<<(FEAT * BBU + 255) / 256, 256, 0, stream>>>(
      bb_w, bbwT, conv2_w, w2h, ff1_w, ff2_w, ta_w, tb_w, whT);
  conv_frame_kernel<<<FRAMES, 128, 0, stream>>>(
      x, conv1_w, conv1_b, w2h, conv2_b, r_in);
  gemm_xz_kernel<<<FRAMES / 16, 256, 0, stream>>>(r_in, bbwT, bb_b, xz);
  cfc_scan_kernel<<<BATCH, 128, 0, stream>>>(
      xz, bb_w, whT, ff1_b, ff2_b, ta_b, tb_b, out_w, out_b, logits, hx_out);
}
